// EdgeModel_2473901163254
// MI455X (gfx1250) — matrix operand order repaired and checked by reading, not compiled, not run on hardware
//
#include <hip/hip_runtime.h>

typedef __attribute__((ext_vector_type(16))) _Float16 v16h;
typedef __attribute__((ext_vector_type(8)))  float    v8f;
typedef __attribute__((ext_vector_type(4)))  float    v4f;   // native vector: OK for nontemporal builtins

#define NODE_NF   32
#define EDGE_NF   32
#define GLOBAL_NF 16
#define HIDDEN_NF 64
#define IN_NF     112

// WMMA f16 A-fragment layout (16x32 MxK, wave32):
//   lane = half*16 + laneM ; row M = laneM
//   element t (0..15):  K = kbase + (t<8 ? t : t+8) + half*8
// -> two contiguous 8-element runs per lane: [kbase+half*8, +8) and [kbase+16+half*8, +8)
__device__ __forceinline__ void loadA32(const float* __restrict__ p, v16h& A) {
    // p points at (row base + half*8) within a 32-float feature block; 16B aligned
    const v4f a0 = __builtin_nontemporal_load((const v4f*)(p));
    const v4f a1 = __builtin_nontemporal_load((const v4f*)(p + 4));
    const v4f a2 = __builtin_nontemporal_load((const v4f*)(p + 16));
    const v4f a3 = __builtin_nontemporal_load((const v4f*)(p + 20));
    A[0]=(_Float16)a0.x; A[1]=(_Float16)a0.y; A[2]=(_Float16)a0.z; A[3]=(_Float16)a0.w;
    A[4]=(_Float16)a1.x; A[5]=(_Float16)a1.y; A[6]=(_Float16)a1.z; A[7]=(_Float16)a1.w;
    A[8]=(_Float16)a2.x; A[9]=(_Float16)a2.y; A[10]=(_Float16)a2.z; A[11]=(_Float16)a2.w;
    A[12]=(_Float16)a3.x; A[13]=(_Float16)a3.y; A[14]=(_Float16)a3.z; A[15]=(_Float16)a3.w;
}

// cap at 256 VGPRs (4 waves/SIMD) -> no s_set_vgpr_msb churn, better occupancy
__global__ __launch_bounds__(256, 4) void edge_mlp_wmma(
    const float* __restrict__ src, const float* __restrict__ dstn,
    const float* __restrict__ ea,  const float* __restrict__ u,
    const int*   __restrict__ batch,
    const float* __restrict__ W1,  const float* __restrict__ b1,
    const float* __restrict__ W2,  const float* __restrict__ b2,
    float* __restrict__ out, int E)
{
    __shared__ _Float16 hbuf[8][16][72];   // per-wave slice, padded stride (144B) vs bank conflicts

    const int lane  = threadIdx.x & 31;
    const int wib   = threadIdx.x >> 5;
    const int half  = lane >> 4;           // 0: lanes 0-15, 1: lanes 16-31
    const int laneM = lane & 15;
    const int waveId = blockIdx.x * (blockDim.x >> 5) + wib;
    const int nwaves = gridDim.x * (blockDim.x >> 5);

    // ---------- persistent weight fragments in VGPRs ----------
    // B-fragment layout (32x16 KxN): lane column n = ntile*16 + laneM,
// element t -> K = kchunk*32 + (t<8 ? t : t+8) + half*8  (matches A-fragment order)
    // Bias trick: virtual K=112 feature == 1.0 in A, == b1[n] in B  -> C init is 0.
    v16h B1[4][4];
    #pragma unroll
    for (int j = 0; j < 4; ++j) {
        const int n = j * 16 + laneM;
        #pragma unroll
        for (int kc = 0; kc < 4; ++kc) {
            v16h f;
            #pragma unroll
            for (int t = 0; t < 16; ++t) {
                const int k = kc * 32 + (t < 8 ? t : t + 8) + half * 8;
                f[t] = (k < IN_NF) ? (_Float16)W1[n * IN_NF + k]
                     : (k == IN_NF) ? (_Float16)b1[n]
                                    : (_Float16)0.0f;
            }
            B1[j][kc] = f;
        }
    }
    v16h B2[2][2];
    #pragma unroll
    for (int j2 = 0; j2 < 2; ++j2) {
        const int n = j2 * 16 + laneM;
        #pragma unroll
        for (int kc = 0; kc < 2; ++kc) {
            v16h f;
            #pragma unroll
            for (int t = 0; t < 16; ++t)
                f[t] = (_Float16)W2[n * HIDDEN_NF + kc * 32 + (t < 8 ? t : t + 8) + half * 8];
            B2[j2][kc] = f;
        }
    }
    float bias2[2];
    #pragma unroll
    for (int j2 = 0; j2 < 2; ++j2) bias2[j2] = b2[j2 * 16 + laneM];

    const int ntiles = (E + 15) >> 4;
    for (int tile = waveId; tile < ntiles; tile += nwaves) {
        int e = tile * 16 + laneM;
        if (e >= E) e = E - 1;             // clamp loads; EXEC stays all-1s for WMMA

        // ---------------- GEMM1: [16,112+bias] x [112+bias,64] ----------------
        v8f C[4] = {};                     // zero -> inline-0 SRC2 eligible

        #pragma unroll
        for (int kc = 0; kc < 4; ++kc) {
            v16h A;
            if (kc == 0)      loadA32(src  + (long)e * 32 + half * 8, A);
            else if (kc == 1) loadA32(dstn + (long)e * 32 + half * 8, A);
            else if (kc == 2) loadA32(ea   + (long)e * 32 + half * 8, A);
            else {
                const int bg = batch[e];
                const float* p = u + bg * GLOBAL_NF + half * 8;   // K=96..111 (u is L2-hot: RT load)
                const v4f a0 = *(const v4f*)(p);
                const v4f a1 = *(const v4f*)(p + 4);
                A[0]=(_Float16)a0.x; A[1]=(_Float16)a0.y; A[2]=(_Float16)a0.z; A[3]=(_Float16)a0.w;
                A[4]=(_Float16)a1.x; A[5]=(_Float16)a1.y; A[6]=(_Float16)a1.z; A[7]=(_Float16)a1.w;
                // K=112 virtual bias feature lives in half=0 lanes at element t=8
                A[8] = half ? (_Float16)0.0f : (_Float16)1.0f;
                #pragma unroll
                for (int t = 9; t < 16; ++t) A[t] = (_Float16)0.0f;  // K=113..127 pad
            }
            #pragma unroll
            for (int j = 0; j < 4; ++j)
                C[j] = __builtin_amdgcn_wmma_f32_16x16x32_f16(
                           false, A, false, B1[j][kc], (short)0, C[j], false, false);
        }

        // ---------------- ReLU + C-layout -> A-layout transpose through LDS ----------------
        #pragma unroll
        for (int j = 0; j < 4; ++j) {
            #pragma unroll
            for (int r = 0; r < 8; ++r) {
                float v = C[j][r];
                v = v > 0.0f ? v : 0.0f;
                hbuf[wib][r + 8 * half][j * 16 + laneM] = (_Float16)v;
            }
        }
        // intra-wave LDS is in-order; wait + clobber stops compiler reordering
        asm volatile("s_wait_dscnt 0" ::: "memory");

        // ---------------- GEMM2: [16,64] x [64,32] ----------------
        v8f C2[2] = {};
        const _Float16* hp = &hbuf[wib][laneM][0];
        #pragma unroll
        for (int kc2 = 0; kc2 < 2; ++kc2) {
            v16h A2;
            #pragma unroll
            for (int t = 0; t < 8; ++t) A2[t]     = hp[kc2 * 32 +      half * 8 + t];
            #pragma unroll
            for (int t = 0; t < 8; ++t) A2[8 + t] = hp[kc2 * 32 + 16 + half * 8 + t];
            #pragma unroll
            for (int j2 = 0; j2 < 2; ++j2)
                C2[j2] = __builtin_amdgcn_wmma_f32_16x16x32_f16(
                             false, A2, false, B2[j2][kc2], (short)0, C2[j2], false, false);
        }
        asm volatile("" ::: "memory");   // keep next-iter LDS stores after these reads

        // ------- store [16,32] tile (bias2 added here); 64B contiguous per half-wave -------
        #pragma unroll
        for (int j2 = 0; j2 < 2; ++j2) {
            #pragma unroll
            for (int r = 0; r < 8; ++r) {
                const int m = tile * 16 + r + 8 * half;
                if (m < E)
                    __builtin_nontemporal_store(C2[j2][r] + bias2[j2],
                                                out + (long)m * 32 + j2 * 16 + laneM);
            }
        }
    }
}

extern "C" void kernel_launch(void* const* d_in, const int* in_sizes, int n_in,
                              void* d_out, int out_size, void* d_ws, size_t ws_size,
                              hipStream_t stream) {
    const float* src   = (const float*)d_in[0];
    const float* dstn  = (const float*)d_in[1];
    const float* ea    = (const float*)d_in[2];
    const float* u     = (const float*)d_in[3];
    const int*   batch = (const int*)d_in[4];
    const float* W1    = (const float*)d_in[5];
    const float* b1    = (const float*)d_in[6];
    const float* W2    = (const float*)d_in[7];
    const float* b2    = (const float*)d_in[8];
    float* out = (float*)d_out;

    const int E = in_sizes[0] / NODE_NF;
    const int ntiles = (E + 15) / 16;
    int blocks = 2048;                          // 16K waves persistent over 250K tiles
    const int blocksNeeded = (ntiles + 7) / 8;  // 8 waves per 256-thread block
    if (blocks > blocksNeeded) blocks = blocksNeeded;
    if (blocks < 1) blocks = 1;

    edge_mlp_wmma<<<blocks, 256, 0, stream>>>(src, dstn, ea, u, batch,
                                              W1, b1, W2, b2, out, E);
}
